// CRNNClassifier_59742995088101
// MI455X (gfx1250) — matrix operand order repaired and checked by reading, not compiled, not run on hardware
//
#include <hip/hip_runtime.h>
#include <hip/hip_bf16.h>

// CDNA5 / gfx1250 ConvLSTM classifier.
// GEMM core: v_wmma_f32_16x16x32_bf16 (bf16 operands, f32 accumulate).
// Data movement: Tensor Data Mover (TENSOR_LOAD_TO_LDS) stages each
// timestep's x tile (80 rows x 64 dwords) into halo-padded LDS rows
// (stride 72 dwords via pad_interval=64dw/pad_amount=8dw), issued one
// timestep ahead and overlapped with the LSTM pointwise phase; drained
// with s_wait_tensorcnt before the next GEMM phase.
//
// Conv1d(x,w_ih)+Conv1d(h,w_hh) fused as one GEMM with K padded 50->96,
// encoded kk = src*48 + k*8 + ic (shift/mask decode; invalid slots carry
// zero weights, so A-side clamped loads are multiplied by 0).
// Layouts from CDNA5 ISA 7.12.2 (wave32):
//   A (16x32 bf16): lane m=l%16; elem j -> K = (j<8 ? b2+j : 16+b2+j-8), b2=8*(lane>=16)
//   B (32x16 bf16): lane col n=l%16; elem j -> K = (j<8 ? b2+j : 16+b2+j-8), b2=8*(lane>=16)
//   D (16x16 f32):  lane col n=l%16; vgpr v -> row m = v + 8*(lane>=16)

typedef __attribute__((ext_vector_type(16))) __bf16        v16bf;
typedef __attribute__((ext_vector_type(8)))  float         v8f;
typedef __attribute__((ext_vector_type(4)))  unsigned int  v4u;
typedef __attribute__((ext_vector_type(8)))  int           v8i;
typedef __attribute__((ext_vector_type(4)))  int           v4i;

#if defined(__has_builtin)
#if __has_builtin(__builtin_amdgcn_tensor_load_to_lds)
#define HAVE_TDM 1
#endif
#endif
#ifndef HAVE_TDM
#define HAVE_TDM 0
#endif

#define CCH 5
#define KW 5
#define LL 64
#define TT 16
#define NB 16384
#define GATES 20
#define BT 16            // batch tile per workgroup
#define RST 72           // padded row stride (floats): halo(4) + 64 + halo(4)
#define MST (CCH*RST)    // 360 floats per batch row (x/h layout [m][ic][RST])

// LDS float offsets
#define XOFF 0
#define XSZ  (BT*MST)          // 5760
#define HOFF (XOFF+XSZ)        // 5760
#define COFF (HOFF+XSZ)        // 11520
#define CSZ  (CCH*BT*LL)       // 5120
#define GOFF (COFF+CSZ)        // 16640
#define GSZ  (GATES*BT*LL)     // 20480
#define BSOFF (GOFF+GSZ)       // 37120
#define TOTF  (BSOFF+32)       // 37152 floats -> 148,608 bytes LDS

__device__ __forceinline__ float sigf(float x) {
    return 1.f / (1.f + __expf(-x));
}
__device__ __forceinline__ float tanh_fast(float x) {
    float e = __expf(-2.f * x);
    return (1.f - e) / (1.f + e);
}

// kk -> LDS offset (floats, relative to x base; h sits XSZ past x, same layout).
// Invalid (ic>4 || k>4) slots clamp to a real in-range element; the matching
// B weight is 0 so the product vanishes.
__device__ __forceinline__ int decode_off(int kk) {
    int src = (kk >= 48) ? 1 : 0;
    int kkr = kk - src * 48;
    int k  = kkr >> 3;
    int ic = kkr & 7;
    if (k  > 4) k  = 4;
    if (ic > 4) ic = 4;
    return src * XSZ + ic * RST + k;   // caller adds m*MST + (l + 2)
}

__device__ __forceinline__ float wval(int kk, int oc,
                                      const float* __restrict__ wih,
                                      const float* __restrict__ whh) {
    int src = (kk >= 48) ? 1 : 0;
    int kkr = kk - src * 48;
    int k  = kkr >> 3;
    int ic = kkr & 7;
    if (k > 4 || ic > 4 || oc >= GATES) return 0.f;
    const float* w = src ? whh : wih;
    return w[(oc * CCH + ic) * KW + k];
}

#if HAVE_TDM
// One 2-D TDM descriptor: 80 rows x 64 dwords from global, landing at
// LDS stride 72 dwords (pad 8 dwords after each 64), halos untouched.
__device__ __forceinline__ void tdm_stage_x(const float* __restrict__ gsrc,
                                            unsigned lds_byte) {
    unsigned long long ga = (unsigned long long)(size_t)gsrc;
    v4u g0 = { 1u,                                   // count=1, user mode
               lds_byte,                             // lds_addr (bytes)
               (unsigned)(ga & 0xFFFFFFFFu),         // global_addr[31:0]
               (unsigned)((ga >> 32) & 0x01FFFFFFu)  // global_addr[56:32]
                   | (2u << 30) };                   // type = 2 ("image")
    v8i g1 = { (int)((2u << 16)      // data_size = 4B
                   | (1u << 20)      // pad_enable
                   | (5u << 22)      // pad_interval: 64 dwords
                   | (7u << 25)),    // pad_amount:   8 dwords
               (int)(64u << 16),     // tensor_dim0 = 64 (low16 @ bits[63:48])
               (int)(80u << 16),     // tensor_dim0 hi=0 | tensor_dim1 = 80
               (int)(64u << 16),     // tensor_dim1 hi=0 | tile_dim0 = 64
               (int)80u,             // tile_dim1 = 80 | tile_dim2 = 0
               64,                   // tensor_dim0_stride = 64 (low32)
               0, 0 };               // stride0 hi | stride1 (unused, 2-D)
    v4i gz4 = { 0, 0, 0, 0 };
    v8i gz8 = { 0, 0, 0, 0, 0, 0, 0, 0 };
    // clang-23 / therock-10.0 arity: (g0, g1, g2, g3, g4, cpol)
    __builtin_amdgcn_tensor_load_to_lds(g0, g1, gz4, gz4, gz8, 0);
}
#endif

__global__ void crnn_convlstm_kernel(const float* __restrict__ x,
                                     const float* __restrict__ wih,
                                     const float* __restrict__ whh,
                                     const float* __restrict__ bih,
                                     const float* __restrict__ bhh,
                                     const float* __restrict__ fcw,
                                     const float* __restrict__ fcb,
                                     const float* __restrict__ baseline,
                                     float* __restrict__ out) {
    extern __shared__ __align__(16) float lds[];
    const int tid  = threadIdx.x;
    const int b0   = blockIdx.x * BT;
    const int lane = tid & 31;
    const int wv   = tid >> 5;          // 8 waves
    const int m    = lane & 15;         // A row (batch) / B,D column index
    const int hi   = lane >> 4;
    const int b2   = hi * 8;

    // ---- init LDS (zeros also establish the conv halos once) ----
    for (int i = tid; i < TOTF; i += 256) lds[i] = 0.f;
    if (tid < GATES) lds[BSOFF + tid] = bih[tid] + bhh[tid];

    // ---- precompute per-lane A-fragment LDS offsets (loop invariant) ----
    int aoff[3][16];
    // ---- precompute B (weight) fragments: 2 N-tiles x 3 K-steps ----
    v16bf bfr[2][3];
#pragma unroll
    for (int s = 0; s < 3; ++s) {
#pragma unroll
        for (int j = 0; j < 16; ++j) {
            int kkA = 32 * s + ((j < 8) ? (b2 + j) : (16 + b2 + (j - 8)));
            aoff[s][j] = decode_off(kkA);
            int kkB = 32 * s + ((j < 8) ? (b2 + j) : (16 + b2 + (j - 8)));
#pragma unroll
            for (int p = 0; p < 2; ++p) {
                bfr[p][s][j] = (__bf16)wval(kkB, m + 16 * p, wih, whh);
            }
        }
    }
    __syncthreads();   // zeros (incl. halos) visible before any x staging

    const float* xtile0 = x + ((size_t)b0) * (CCH * LL);
    const size_t xstep  = (size_t)NB * (CCH * LL);   // floats per timestep

#if HAVE_TDM
    const unsigned xlds = (unsigned)(size_t)&lds[XOFF] + 16u; // skip 4-dword halo
    if (wv == 0) tdm_stage_x(xtile0, xlds);          // prologue: stage x(0)
    __builtin_amdgcn_s_wait_tensorcnt(0);
    __syncthreads();
#else
    {
        const float4* xg = (const float4*)xtile0;
#pragma unroll
        for (int q = 0; q < 5; ++q) {
            int f4 = tid + q * 256;
            int mm = f4 / 80, r = f4 % 80, ic = r / 16, l4 = r % 16;
            float4 v = xg[(mm * CCH + ic) * 16 + l4];
            *(float4*)&lds[XOFF + mm * MST + ic * RST + 4 + l4 * 4] = v;
        }
    }
    __syncthreads();
#endif

    // ---- time recurrence ----
    for (int t = 0; t < TT; ++t) {
        // ---- phase A: gates = [x|h] * W via WMMA, each wave owns 8 l's ----
#pragma unroll 1
        for (int il = 0; il < 8; ++il) {
            int l    = wv * 8 + il;
            int base = XOFF + m * MST + l + 2;   // pos = l + k - 2 + halo(4)
            v8f acc0 = {};
            v8f acc1 = {};
#pragma unroll
            for (int s = 0; s < 3; ++s) {
                v16bf af;
#pragma unroll
                for (int j = 0; j < 16; ++j)
                    af[j] = (__bf16)lds[base + aoff[s][j]];
                acc0 = __builtin_amdgcn_wmma_f32_16x16x32_bf16(
                    false, af, false, bfr[0][s], (short)0, acc0, false, false);
                acc1 = __builtin_amdgcn_wmma_f32_16x16x32_bf16(
                    false, af, false, bfr[1][s], (short)0, acc1, false, false);
            }
            // D layout: lane col n = lane%16; vgpr v -> row m = v + 8*hi
            int n = m;
#pragma unroll
            for (int v = 0; v < 8; ++v) {
                int mm = v + b2;
                lds[GOFF + (n * BT + mm) * LL + l] = acc0[v];
            }
            if (n < GATES - 16) {               // N-tile 1: oc 16..19 only
#pragma unroll
                for (int v = 0; v < 8; ++v) {
                    int mm = v + b2;
                    lds[GOFF + ((n + 16) * BT + mm) * LL + l] = acc1[v];
                }
            }
        }
        __syncthreads();   // all of x(t),h consumed; gates complete

        // ---- stage x(t+1): async TDM overlapped with pointwise phase B ----
        if (t + 1 < TT) {
#if HAVE_TDM
            if (wv == 0) tdm_stage_x(xtile0 + (size_t)(t + 1) * xstep, xlds);
#else
            const float4* xg = (const float4*)(xtile0 + (size_t)(t + 1) * xstep);
#pragma unroll
            for (int q = 0; q < 5; ++q) {
                int f4 = tid + q * 256;
                int mm = f4 / 80, r = f4 % 80, ic = r / 16, l4 = r % 16;
                float4 v = xg[(mm * CCH + ic) * 16 + l4];
                *(float4*)&lds[XOFF + mm * MST + ic * RST + 4 + l4 * 4] = v;
                if (t + 2 < TT)
                    __builtin_prefetch((const void*)(xg + xstep / 4 +
                                                     (mm * CCH + ic) * 16 + l4), 0, 1);
            }
#endif
        }

        // ---- phase B: LSTM pointwise update of c and h ----
#pragma unroll 1
        for (int idx = tid; idx < CCH * BT * LL; idx += 256) {
            int l  = idx & 63;
            int r  = idx >> 6;
            int mm = r & 15;
            int ch = r >> 4;
            float gi = lds[GOFF + (ch        * BT + mm) * LL + l] + lds[BSOFF + ch];
            float gf = lds[GOFF + ((5  + ch) * BT + mm) * LL + l] + lds[BSOFF + 5 + ch];
            float gg = lds[GOFF + ((10 + ch) * BT + mm) * LL + l] + lds[BSOFF + 10 + ch];
            float go = lds[GOFF + ((15 + ch) * BT + mm) * LL + l] + lds[BSOFF + 15 + ch];
            float ii  = sigf(gi);
            float ff  = sigf(gf);
            float gg2 = tanh_fast(gg);
            float oo  = sigf(go);
            int cix = COFF + (ch * BT + mm) * LL + l;
            float c = ff * lds[cix] + ii * gg2;
            lds[cix] = c;
            lds[HOFF + mm * MST + ch * RST + 4 + l] = oo * tanh_fast(c);
        }

#if HAVE_TDM
        __builtin_amdgcn_s_wait_tensorcnt(0);   // x(t+1) resident (no-op if none)
#endif
        __syncthreads();
    }

    // ---- fc over L, sigmoid, noisy-OR across channels ----
    if (tid < BT) {
        float pbase = sigf(baseline[0]);
        float prod  = 1.f - pbase;
        float fb    = fcb[0];
#pragma unroll 1
        for (int ch = 0; ch < CCH; ++ch) {
            float s = fb;
            const float* hp = &lds[HOFF + tid * MST + ch * RST + 4];
#pragma unroll 1
            for (int l = 0; l < LL; ++l) s += hp[l] * fcw[l];
            prod *= (1.f - sigf(s));
        }
        out[b0 + tid] = 1.f - prod;
    }
}

extern "C" void kernel_launch(void* const* d_in, const int* in_sizes, int n_in,
                              void* d_out, int out_size, void* d_ws, size_t ws_size,
                              hipStream_t stream) {
    (void)in_sizes; (void)n_in; (void)d_ws; (void)ws_size; (void)out_size;
    const float* x        = (const float*)d_in[0];
    const float* w_ih     = (const float*)d_in[1];
    const float* w_hh     = (const float*)d_in[2];
    const float* b_ih     = (const float*)d_in[3];
    const float* b_hh     = (const float*)d_in[4];
    const float* fc_w     = (const float*)d_in[5];
    const float* fc_b     = (const float*)d_in[6];
    const float* baseline = (const float*)d_in[7];
    float* out            = (float*)d_out;

    dim3 grid(NB / BT);          // 1024 workgroups (16-batch tiles)
    dim3 block(256);             // 8 wave32 waves
    size_t smem = (size_t)TOTF * sizeof(float);  // ~148.6 KB -> 2 WGs per 320 KB WGP
    crnn_convlstm_kernel<<<grid, block, smem, stream>>>(
        x, w_ih, w_hh, b_ih, b_hh, fc_w, fc_b, baseline, out);
}
